// PointNet2_26242250179269
// MI455X (gfx1250) — compile-verified, matrix operand order checked by reading, not run on hardware
//
#include <hip/hip_runtime.h>
#include <cstdint>
#include <cstddef>

// ============================ types ============================
typedef _Float16 f16;
typedef __attribute__((ext_vector_type(16))) _Float16 v16h;
typedef __attribute__((ext_vector_type(8)))  float    v8f;

#define BN_EPS 1e-5f

// ===================== WMMA GEMM + BN + ReLU =====================
// out[M,N] = act( BN( A[M,ldA(>=K)] @ W[N,K]^T + bias ) ) stored f16 (stride N).
// A is f16 with row stride ldA (multiple of 32, zero-padded cols K..ldA).
// M is assumed "mostly" a multiple of 64 (guarded block-uniformly otherwise).
// 256 threads = 8 wave32 arranged 4 (M) x 2 (N); each wave owns WN 16x16
// C tiles in N (one A fragment reused WN times). Block tile: 64 x (32*WN).
template <int WN>
__global__ __launch_bounds__(256) void gemm_bn_relu_kernel(
    const f16* __restrict__ A, const float* __restrict__ W,
    const float* __restrict__ bias, const float* __restrict__ g,
    const float* __restrict__ be, const float* __restrict__ rm,
    const float* __restrict__ rv,
    int M, int K, int N, int ldA, int use_bn, int use_relu,
    f16* __restrict__ out)
{
    constexpr int BN_ = 32 * WN;     // 64 (WN=2) or 128 (WN=4)
    constexpr int REP = BN_ / 64;    // W-tile row passes per thread

    __shared__ __align__(16) f16 As[64][32];
    __shared__ __align__(16) f16 Ws[BN_][32];

    const int tid  = threadIdx.x;
    const int lane = tid & 31;
    const int wave = tid >> 5;
    const int h    = lane >> 4;      // half-wave group
    const int ln   = lane & 15;
    const int wm   = wave >> 1;
    const int wn0  = (wave & 1) * WN;

    const int mBase = blockIdx.y * 64;
    const int nBase = blockIdx.x * BN_;

    const int r = tid >> 2;          // staging row 0..63
    const int c = (tid & 3) * 8;     // staging col 0,8,16,24 (16B aligned)

    v8f acc[WN] = {};

    const bool mFull    = (mBase + 64 <= M);
    const bool kAligned = ((K & 3) == 0);

    for (int kb = 0; kb < ldA; kb += 32) {
        // ---- stage A: one b128 load + one ds b128 store (pad cols are zero)
        {
            const int m = mBase + r;
            if (mFull || m < M) {
                *(uint4*)&As[r][c] = *(const uint4*)(A + (size_t)m * ldA + kb + c);
            } else {
                const uint4 z = {0u, 0u, 0u, 0u};
                *(uint4*)&As[r][c] = z;
            }
        }
        // ---- stage W (f32 -> f16)
#pragma unroll
        for (int rep = 0; rep < REP; ++rep) {
            const int rw = rep * 64 + r;
            const int n  = nBase + rw;
            union { uint4 u; f16 a[8]; } tmp;
            if (kAligned && n < N && (kb + c + 8) <= K) {
                const float4 w0 = *(const float4*)(W + (size_t)n * K + kb + c);
                const float4 w1 = *(const float4*)(W + (size_t)n * K + kb + c + 4);
                tmp.a[0] = (f16)w0.x; tmp.a[1] = (f16)w0.y;
                tmp.a[2] = (f16)w0.z; tmp.a[3] = (f16)w0.w;
                tmp.a[4] = (f16)w1.x; tmp.a[5] = (f16)w1.y;
                tmp.a[6] = (f16)w1.z; tmp.a[7] = (f16)w1.w;
            } else {
                // branchless clamped fallback (odd-K layers / edges)
                const int ncl = (n < N) ? n : (N - 1);
#pragma unroll
                for (int q = 0; q < 8; ++q) {
                    const int k   = kb + c + q;
                    const int kcl = (k < K) ? k : (K - 1);
                    const float wv = W[(size_t)ncl * K + kcl];
                    tmp.a[q] = (n < N && k < K) ? (f16)wv : (f16)0.f;
                }
            }
            *(uint4*)&Ws[rw][c] = tmp.u;
        }
        __syncthreads();

        // ---- fragments: lane half-index i -> K = 8h+i (i<8), 16+8h+(i-8)
        union FragU { v16h v; uint4 u[2]; } af, bf;
        const int mr = wm * 16 + ln;
        af.u[0] = *(const uint4*)&As[mr][8 * h];
        af.u[1] = *(const uint4*)&As[mr][16 + 8 * h];
#pragma unroll
        for (int t = 0; t < WN; ++t) {
            const int nr = (wn0 + t) * 16 + ln;
            bf.u[0] = *(const uint4*)&Ws[nr][8 * h];
            bf.u[1] = *(const uint4*)&Ws[nr][16 + 8 * h];
            acc[t] = __builtin_amdgcn_wmma_f32_16x16x32_f16(
                         false, af.v, false, bf.v, (short)0, acc[t], false, false);
        }
        __syncthreads();
    }

    // ---- fused epilogue: y = acc*alpha + beta ; relu ; f16 store (stride N)
#pragma unroll
    for (int t = 0; t < WN; ++t) {
        const int n = nBase + (wn0 + t) * 16 + ln;
        if (n >= N) continue;
        float alpha = 1.f;
        float beta  = bias ? bias[n] : 0.f;
        if (use_bn) {
            const float sc = g[n] * rsqrtf(rv[n] + BN_EPS);
            alpha = sc;
            beta  = (bias[n] - rm[n]) * sc + be[n];
        }
#pragma unroll
        for (int j = 0; j < 8; ++j) {
            const int m = mBase + wm * 16 + 8 * h + j;
            if (m < M) {
                float y = acc[t][j] * alpha + beta;
                if (use_relu) y = fmaxf(y, 0.f);
                out[(size_t)m * N + n] = (f16)y;
            }
        }
    }
}

// ===================== farthest point sampling =====================
// One block (256 threads) per batch element. Matches jax scan semantics:
// out[0] = 0, out[i] = argmax(dist after i min-updates), first-index ties.
__global__ __launch_bounds__(256) void fps_kernel(
    const float* __restrict__ xyz, int N, int npoint, int* __restrict__ out)
{
    __shared__ float dist[4096];
    __shared__ float redV[256];
    __shared__ int   redI[256];
    __shared__ int   curS;

    const int b   = blockIdx.x;
    const int tid = threadIdx.x;
    const float* p = xyz + (size_t)b * N * 3;
    int* o = out + (size_t)b * npoint;

    for (int n = tid; n < N; n += 256) dist[n] = 1e10f;
    if (tid == 0) { o[0] = 0; curS = 0; }
    __syncthreads();

    for (int it = 1; it < npoint; ++it) {
        const int cur = curS;
        const float cx = p[cur * 3 + 0];
        const float cy = p[cur * 3 + 1];
        const float cz = p[cur * 3 + 2];
        float bestV = -1.f; int bestI = 0x7fffffff;
        for (int n = tid; n < N; n += 256) {
            const float dx = p[n * 3 + 0] - cx;
            const float dy = p[n * 3 + 1] - cy;
            const float dz = p[n * 3 + 2] - cz;
            const float d  = dx * dx + dy * dy + dz * dz;
            const float dn = fminf(dist[n], d);
            dist[n] = dn;
            if (dn > bestV || (dn == bestV && n < bestI)) { bestV = dn; bestI = n; }
        }
        redV[tid] = bestV; redI[tid] = bestI;
        __syncthreads();
        for (int s = 128; s > 0; s >>= 1) {
            if (tid < s) {
                const float v2 = redV[tid + s]; const int i2 = redI[tid + s];
                if (v2 > redV[tid] || (v2 == redV[tid] && i2 < redI[tid])) {
                    redV[tid] = v2; redI[tid] = i2;
                }
            }
            __syncthreads();
        }
        if (tid == 0) { curS = redI[0]; o[it] = redI[0]; }
        __syncthreads();
    }
}

// ===================== ball query =====================
// One thread per centroid. Ascending scan == "sort indices, take first
// nsample" in the reference; pad with first hit; N-1 if no hit.
__global__ void ball_query_kernel(
    const float* __restrict__ xyz, const float* __restrict__ new_xyz,
    int N, int S, int ns, float r2, int* __restrict__ out, int total)
{
    const int t = blockIdx.x * blockDim.x + threadIdx.x;
    if (t >= total) return;
    const int b = t / S;
    const float* q = new_xyz + (size_t)t * 3;
    const float* p = xyz + (size_t)b * N * 3;
    int* o = out + (size_t)t * ns;
    const float qx = q[0], qy = q[1], qz = q[2];
    int cnt = 0, first = N - 1;
    for (int n = 0; n < N && cnt < ns; ++n) {
        const float dx = p[n * 3 + 0] - qx;
        const float dy = p[n * 3 + 1] - qy;
        const float dz = p[n * 3 + 2] - qz;
        if (dx * dx + dy * dy + dz * dz <= r2) {
            if (cnt == 0) first = n;
            o[cnt++] = n;
        }
    }
    for (int j = cnt; j < ns; ++j) o[j] = first;
}

// ===================== gather xyz by index =====================
__global__ void gather3_kernel(const float* __restrict__ xyz,
                               const int* __restrict__ idx,
                               int S, int N, float* __restrict__ out, int total)
{
    const int t = blockIdx.x * blockDim.x + threadIdx.x;
    if (t >= total) return;                // total = B*S*3
    const int r = t / 3, c = t - r * 3;
    const int b = r / S;
    out[t] = xyz[((size_t)b * N + idx[r]) * 3 + c];
}

// ===================== grouping (gather + center + concat, padded rows) =====================
// out[b,s,j, 0:Ct] = concat(xyz[idx]-new_xyz, points[idx]); cols Ct..ld = 0
__global__ void group_kernel(
    const float* __restrict__ xyz, const float* __restrict__ new_xyz,
    const int* __restrict__ idx,
    const f16* __restrict__ pts, const float* __restrict__ pts_f32, int C,
    int N, int S, int ns, int ld, f16* __restrict__ out, long long total)
{
    const int Ct = 3 + C;
    for (long long t = (long long)blockIdx.x * blockDim.x + threadIdx.x;
         t < total; t += (long long)gridDim.x * blockDim.x) {
        const long long row = t / ld;
        const int c  = (int)(t - row * ld);
        if (c >= Ct) { out[t] = (f16)0.f; continue; }
        const long long bs = row / ns;
        const int b  = (int)(bs / S);
        const int pid = idx[row];
        float v;
        if (c < 3) {
            v = xyz[((size_t)b * N + pid) * 3 + c] - new_xyz[(size_t)bs * 3 + c];
        } else {
            const int cc = c - 3;
            v = pts_f32 ? pts_f32[((size_t)b * N + pid) * C + cc]
                        : (float)pts[((size_t)b * N + pid) * C + cc];
        }
        out[t] = (f16)v;
    }
}

// ===================== max-pool over samples =====================
__global__ void maxpool_kernel(const f16* __restrict__ in, int rows, int ns,
                               int C, f16* __restrict__ out, long long total)
{
    for (long long t = (long long)blockIdx.x * blockDim.x + threadIdx.x;
         t < total; t += (long long)gridDim.x * blockDim.x) {
        const long long r = t / C;
        const int c = (int)(t - r * C);
        float m = -3.0e38f;
        for (int j = 0; j < ns; ++j)
            m = fmaxf(m, (float)in[((size_t)r * ns + j) * C + c]);
        out[t] = (f16)m;
    }
}

// ===================== SA3 group_all concat: [xyz_f32 | feat_f16], padded =====================
__global__ void concat_xyz_feat_kernel(const float* __restrict__ xyz,
                                       const f16* __restrict__ feat, int C,
                                       int ld, f16* __restrict__ out,
                                       long long total)
{
    const int Ct = 3 + C;
    for (long long t = (long long)blockIdx.x * blockDim.x + threadIdx.x;
         t < total; t += (long long)gridDim.x * blockDim.x) {
        const long long row = t / ld;
        const int c = (int)(t - row * ld);
        if (c >= Ct) { out[t] = (f16)0.f; continue; }
        out[t] = (c < 3) ? (f16)xyz[row * 3 + c] : feat[(size_t)row * C + (c - 3)];
    }
}

// ===================== FP (S==1): broadcast concat, padded =====================
__global__ void bcast_concat_kernel(const f16* __restrict__ p1, int C1,
                                    const f16* __restrict__ p2, int C2,
                                    int Nq, int ld, f16* __restrict__ out,
                                    long long total)
{
    const int Ct = C1 + C2;
    for (long long t = (long long)blockIdx.x * blockDim.x + threadIdx.x;
         t < total; t += (long long)gridDim.x * blockDim.x) {
        const long long row = t / ld;
        const int c = (int)(t - row * ld);
        if (c >= Ct) { out[t] = (f16)0.f; continue; }
        const int b = (int)(row / Nq);
        out[t] = (c < C1) ? p1[(size_t)row * C1 + c]
                          : p2[(size_t)b * C2 + (c - C1)];
    }
}

// ===================== FP: 3-NN inverse-distance interp + concat, padded =====================
// Stable top-3 (strict <) == lax.top_k tie behavior. w = 1/(d+1e-8), normalized.
__global__ void fp_interp_kernel(
    const float* __restrict__ xyz1, const float* __restrict__ xyz2,
    const f16* __restrict__ p1, const float* __restrict__ p1f32, int C1,
    const f16* __restrict__ p2, int C2, int Nq, int S, int ld,
    int total, f16* __restrict__ out)
{
    const int t = blockIdx.x * blockDim.x + threadIdx.x;
    if (t >= total) return;                 // total = B*Nq
    const int b = t / Nq;
    const float* q  = xyz1 + (size_t)t * 3;
    const float* c2 = xyz2 + (size_t)b * S * 3;
    const float qx = q[0], qy = q[1], qz = q[2];

    float d0 = 1e30f, d1 = 1e30f, d2v = 1e30f;
    int   i0 = 0,     i1 = 0,     i2 = 0;
    for (int s = 0; s < S; ++s) {
        const float dx = qx - c2[s * 3 + 0];
        const float dy = qy - c2[s * 3 + 1];
        const float dz = qz - c2[s * 3 + 2];
        const float d  = dx * dx + dy * dy + dz * dz;
        if (d < d0)      { d2v = d1; i2 = i1; d1 = d0; i1 = i0; d0 = d; i0 = s; }
        else if (d < d1) { d2v = d1; i2 = i1; d1 = d;  i1 = s; }
        else if (d < d2v){ d2v = d;  i2 = s; }
    }
    float w0 = 1.f / (d0 + 1e-8f);
    float w1 = 1.f / (d1 + 1e-8f);
    float w2 = 1.f / (d2v + 1e-8f);
    const float inv = 1.f / (w0 + w1 + w2);
    w0 *= inv; w1 *= inv; w2 *= inv;

    f16* o = out + (size_t)t * ld;
    for (int c = 0; c < C1; ++c)
        o[c] = p1f32 ? (f16)p1f32[(size_t)t * C1 + c] : p1[(size_t)t * C1 + c];
    const size_t base2 = (size_t)b * S * C2;
    for (int c = 0; c < C2; ++c) {
        const float v = w0 * (float)p2[base2 + (size_t)i0 * C2 + c]
                      + w1 * (float)p2[base2 + (size_t)i1 * C2 + c]
                      + w2 * (float)p2[base2 + (size_t)i2 * C2 + c];
        o[C1 + c] = (f16)v;
    }
    for (int c = C1 + C2; c < ld; ++c) o[c] = (f16)0.f;
}

// ===================== log-softmax over 13 classes =====================
__global__ void logsoftmax_kernel(const f16* __restrict__ logits, int rows,
                                  int C, float* __restrict__ out)
{
    const int t = blockIdx.x * blockDim.x + threadIdx.x;
    if (t >= rows) return;
    const f16* x = logits + (size_t)t * C;
    float m = -3.0e38f;
    for (int c = 0; c < C; ++c) m = fmaxf(m, (float)x[c]);
    float s = 0.f;
    for (int c = 0; c < C; ++c) s += expf((float)x[c] - m);
    const float l = logf(s);
    float* o = out + (size_t)t * C;
    for (int c = 0; c < C; ++c) o[c] = (float)x[c] - m - l;
}

// ============================ host launcher ============================
struct ConvBN { const float *b, *be, *g, *rm, *rv, *w; };

static ConvBN takeCB(void* const* d_in, int& i) {
    // JAX sorted-dict leaf order within a ConvBN dict: b, be, g, rm, rv, w
    ConvBN p;
    p.b  = (const float*)d_in[i++];
    p.be = (const float*)d_in[i++];
    p.g  = (const float*)d_in[i++];
    p.rm = (const float*)d_in[i++];
    p.rv = (const float*)d_in[i++];
    p.w  = (const float*)d_in[i++];
    return p;
}

static inline unsigned cdiv(long long a, long long b) { return (unsigned)((a + b - 1) / b); }

extern "C" void kernel_launch(void* const* d_in, const int* in_sizes, int n_in,
                              void* d_out, int out_size, void* d_ws, size_t ws_size,
                              hipStream_t stream)
{
    (void)in_sizes; (void)n_in; (void)out_size; (void)ws_size;

    // ---- decode params (sorted-key flatten: fp1,fp2,fp3,head1,head2,sa1,sa2,sa3, then xyz)
    int i = 0;
    ConvBN fp1[3]; for (int l = 0; l < 3; ++l) fp1[l] = takeCB(d_in, i);
    ConvBN fp2[2]; for (int l = 0; l < 2; ++l) fp2[l] = takeCB(d_in, i);
    ConvBN fp3[2]; for (int l = 0; l < 2; ++l) fp3[l] = takeCB(d_in, i);
    ConvBN head1 = takeCB(d_in, i);
    const float* head2_b = (const float*)d_in[i++];
    const float* head2_w = (const float*)d_in[i++];
    ConvBN sa1[3]; for (int l = 0; l < 3; ++l) sa1[l] = takeCB(d_in, i);
    ConvBN sa2[3]; for (int l = 0; l < 3; ++l) sa2[l] = takeCB(d_in, i);
    ConvBN sa3[3]; for (int l = 0; l < 3; ++l) sa3[l] = takeCB(d_in, i);
    const float* xyz = (const float*)d_in[i++];

    const int B = 16, N0 = 4096;

    // ---- workspace carve (256B aligned)
    char* wsp = (char*)d_ws;
    auto alloc = [&](size_t bytes) -> void* {
        void* p = (void*)wsp;
        wsp += (bytes + 255) & ~(size_t)255;
        return p;
    };
    int*   fps1  = (int*)  alloc((size_t)B * 512 * 4);
    float* nx1   = (float*)alloc((size_t)B * 512 * 3 * 4);
    int*   ball1 = (int*)  alloc((size_t)B * 512 * 32 * 4);
    int*   fps2  = (int*)  alloc((size_t)B * 128 * 4);
    float* nx2   = (float*)alloc((size_t)B * 128 * 3 * 4);
    int*   ball2 = (int*)  alloc((size_t)B * 128 * 64 * 4);
    f16*   l1p   = (f16*)  alloc((size_t)B * 512 * 128 * 2);
    f16*   l2p   = (f16*)  alloc((size_t)B * 128 * 256 * 2);
    f16*   l3p   = (f16*)  alloc((size_t)B * 1024 * 2);
    f16*   g1    = (f16*)  alloc((size_t)B * 512 * 32 * 32 * 2);   // ld=32
    const size_t ACT_BYTES = (size_t)33554432 * 2;   // 33.55M f16 = 64 MiB
    f16*   actA  = (f16*)  alloc(ACT_BYTES);
    f16*   actB  = (f16*)  alloc(ACT_BYTES);

    auto gemm = [&](const f16* A, int ldA, const ConvBN& p, int M, int K, int N,
                    f16* O, int bn, int relu) {
        if (N >= 128) {
            dim3 grid(cdiv(N, 128), cdiv(M, 64));
            gemm_bn_relu_kernel<4><<<grid, 256, 0, stream>>>(
                A, p.w, p.b, p.g, p.be, p.rm, p.rv, M, K, N, ldA, bn, relu, O);
        } else {
            dim3 grid(cdiv(N, 64), cdiv(M, 64));
            gemm_bn_relu_kernel<2><<<grid, 256, 0, stream>>>(
                A, p.w, p.b, p.g, p.be, p.rm, p.rv, M, K, N, ldA, bn, relu, O);
        }
    };

    // ================= SA1: 4096 -> 512 centroids, ns=32, [6->64->64->128] =================
    fps_kernel<<<B, 256, 0, stream>>>(xyz, N0, 512, fps1);
    gather3_kernel<<<cdiv((long long)B * 512 * 3, 256), 256, 0, stream>>>(
        xyz, fps1, 512, N0, nx1, B * 512 * 3);
    ball_query_kernel<<<cdiv(B * 512, 256), 256, 0, stream>>>(
        xyz, nx1, N0, 512, 32, 0.04f, ball1, B * 512);
    {
        long long tot = (long long)B * 512 * 32 * 32;   // ld=32 (Ct=6)
        group_kernel<<<cdiv(tot, 256), 256, 0, stream>>>(
            xyz, nx1, ball1, nullptr, xyz, 3, N0, 512, 32, 32, g1, tot);
    }
    const int M1 = B * 512 * 32;
    gemm(g1,   32, sa1[0], M1, 6,   64,  actA, 1, 1);
    gemm(actA, 64, sa1[1], M1, 64,  64,  actB, 1, 1);
    gemm(actB, 64, sa1[2], M1, 64,  128, actA, 1, 1);
    {
        long long tot = (long long)B * 512 * 128;
        maxpool_kernel<<<cdiv(tot, 256), 256, 0, stream>>>(actA, B * 512, 32, 128, l1p, tot);
    }

    // ================= SA2: 512 -> 128 centroids, ns=64, [131->128->128->256] =================
    fps_kernel<<<B, 256, 0, stream>>>(nx1, 512, 128, fps2);
    gather3_kernel<<<cdiv((long long)B * 128 * 3, 256), 256, 0, stream>>>(
        nx1, fps2, 128, 512, nx2, B * 128 * 3);
    ball_query_kernel<<<cdiv(B * 128, 256), 256, 0, stream>>>(
        nx1, nx2, 512, 128, 64, 0.16f, ball2, B * 128);
    {
        long long tot = (long long)B * 128 * 64 * 160;  // ld=160 (Ct=131)
        group_kernel<<<cdiv(tot, 256), 256, 0, stream>>>(
            nx1, nx2, ball2, l1p, nullptr, 128, 512, 128, 64, 160, actA, tot);
    }
    const int M2 = B * 128 * 64;
    gemm(actA, 160, sa2[0], M2, 131, 128, actB, 1, 1);
    gemm(actB, 128, sa2[1], M2, 128, 128, actA, 1, 1);
    gemm(actA, 128, sa2[2], M2, 128, 256, actB, 1, 1);
    {
        long long tot = (long long)B * 128 * 256;
        maxpool_kernel<<<cdiv(tot, 256), 256, 0, stream>>>(actB, B * 128, 64, 256, l2p, tot);
    }

    // ================= SA3: group_all, [259->256->512->1024] =================
    {
        long long tot = (long long)B * 128 * 288;       // ld=288 (Ct=259)
        concat_xyz_feat_kernel<<<cdiv(tot, 256), 256, 0, stream>>>(
            nx2, l2p, 256, 288, actA, tot);
    }
    const int M3 = B * 128;
    gemm(actA, 288, sa3[0], M3, 259, 256,  actB, 1, 1);
    gemm(actB, 256, sa3[1], M3, 256, 512,  actA, 1, 1);
    gemm(actA, 512, sa3[2], M3, 512, 1024, actB, 1, 1);
    {
        long long tot = (long long)B * 1024;
        maxpool_kernel<<<cdiv(tot, 256), 256, 0, stream>>>(actB, B, 128, 1024, l3p, tot);
    }

    // ================= FP3 (S==1 broadcast): [1280->256->256] =================
    {
        long long tot = (long long)B * 128 * 1280;      // ld=1280 (Ct=1280)
        bcast_concat_kernel<<<cdiv(tot, 256), 256, 0, stream>>>(
            l2p, 256, l3p, 1024, 128, 1280, actA, tot);
    }
    gemm(actA, 1280, fp3[0], M3, 1280, 256, actB, 1, 1);
    gemm(actB, 256,  fp3[1], M3, 256,  256, actA, 1, 1);   // actA = refined l2 features

    // ================= FP2: 128 -> 512, [384->256->128] =================
    fp_interp_kernel<<<cdiv(B * 512, 128), 128, 0, stream>>>(
        nx1, nx2, l1p, nullptr, 128, actA, 256, 512, 128, 384, B * 512, actB);
    gemm(actB, 384, fp2[0], B * 512, 384, 256, actA, 1, 1);
    gemm(actA, 256, fp2[1], B * 512, 256, 128, actB, 1, 1); // actB = refined l1 features

    // ================= FP1: 512 -> 4096, [131->128->128->128] =================
    fp_interp_kernel<<<cdiv(B * N0, 128), 128, 0, stream>>>(
        xyz, nx1, nullptr, xyz, 3, actB, 128, N0, 512, 160, B * N0, actA);
    const int M0 = B * N0;
    gemm(actA, 160, fp1[0], M0, 131, 128, actB, 1, 1);
    gemm(actB, 128, fp1[1], M0, 128, 128, actA, 1, 1);
    gemm(actA, 128, fp1[2], M0, 128, 128, actB, 1, 1);

    // ================= head + log-softmax =================
    gemm(actB, 128, head1, M0, 128, 128, actA, 1, 1);
    ConvBN h2; h2.b = head2_b; h2.be = nullptr; h2.g = nullptr;
    h2.rm = nullptr; h2.rv = nullptr; h2.w = head2_w;
    gemm(actA, 128, h2, M0, 128, 13, actB, 0, 0);
    logsoftmax_kernel<<<cdiv(M0, 256), 256, 0, stream>>>(actB, M0, 13, (float*)d_out);
}
